// AttGNN_67095979098342
// MI455X (gfx1250) — matrix operand order repaired and checked by reading, not compiled, not run on hardware
//
#include <hip/hip_runtime.h>
#include <cstddef>
#include <cstdint>

// ---------------- problem constants (match reference) ----------------
#define B_    4
#define V_    2048
#define FIN_  256
#define HID_  256
#define COND_ 64
#define ENC_  128
#define HEADS_ 8
#define DH_   32
#define FF_   1024
#define NL_   3
#define M_    (B_*V_)

typedef unsigned short u16;
typedef unsigned int   u32;
typedef __attribute__((ext_vector_type(4))) u32    u32x4;
typedef __attribute__((ext_vector_type(2))) u32    u32x2;
typedef __attribute__((ext_vector_type(16))) __bf16 v16bf;
typedef __attribute__((ext_vector_type(8)))  float  v8f;

union BF16x16 { v16bf v; u16 u[16]; u32x4 q[2]; };

__device__ inline u16 f2bf(float f) {
  u32 u = __float_as_uint(f);
  u = u + 0x7FFFu + ((u >> 16) & 1u);   // round-to-nearest-even
  return (u16)(u >> 16);
}

__device__ inline v8f wmma_bf16(const BF16x16& a, const BF16x16& b, v8f c) {
  // D = A(16x32 bf16) * B(32x16 bf16) + C(16x16 f32)
  return __builtin_amdgcn_wmma_f32_16x16x32_bf16(false, a.v, false, b.v,
                                                 (short)0, c, false, false);
}

// =====================================================================
// Generic WMMA GEMM: C[M,N] = op(A_bf16[M,K] @ W_bf16[K,N] + bias) + residual
// Requirements: K%32==0, N%128==0, rows 16B-aligned (K%8==0, N%8==0).
// Block: 256 threads = 8 waves in a 2(M)x4(N) grid; block tile 64x128.
// =====================================================================
__global__ __launch_bounds__(256)
void k_gemm(const u16* __restrict__ A, const u16* __restrict__ W,
            const float* __restrict__ bias, const float* __restrict__ residual,
            float* __restrict__ C, int Mdim, int Ndim, int Kdim, int relu)
{
  __shared__ u16 As[64][40];    // row-major [m][k], stride 80B (16B multiple)
  __shared__ u16 Bst[128][40];  // transposed [n][k]

  const int tid  = threadIdx.x;
  const int wave = tid >> 5, lane = tid & 31;
  const int wm = wave >> 2, wn = wave & 3;
  const int lm = lane & 15, lh = lane >> 4;
  const int m0 = blockIdx.y * 64;
  const int n0 = blockIdx.x * 128;

  v8f acc[2][2];
  #pragma unroll
  for (int a = 0; a < 2; a++)
    #pragma unroll
    for (int b = 0; b < 2; b++)
      #pragma unroll
      for (int r = 0; r < 8; r++) acc[a][b][r] = 0.0f;

  for (int k0 = 0; k0 < Kdim; k0 += 32) {
    __syncthreads();
    { // A tile: 64x32 bf16, 8 elems (16B) per thread
      int idx = tid * 8;
      int r = idx >> 5, c = idx & 31;
      int gr = m0 + r;
      u32x4 val = {0u, 0u, 0u, 0u};
      if (gr < Mdim)
        val = *reinterpret_cast<const u32x4*>(A + (size_t)gr * Kdim + k0 + c);
      *reinterpret_cast<u32x4*>(&As[r][c]) = val;
    }
    { // B tile: 32x128 bf16 loaded row-major, stored transposed
      int idx = tid * 16;
      int r = idx >> 7, c = idx & 127;
      const u16* wp = W + (size_t)(k0 + r) * Ndim + n0 + c;
      u32x4 w0 = *reinterpret_cast<const u32x4*>(wp);
      u32x4 w1 = *reinterpret_cast<const u32x4*>(wp + 8);
      const u16* e0 = reinterpret_cast<const u16*>(&w0);
      const u16* e1 = reinterpret_cast<const u16*>(&w1);
      #pragma unroll
      for (int i = 0; i < 8; i++) Bst[c + i][r]     = e0[i];
      #pragma unroll
      for (int i = 0; i < 8; i++) Bst[c + 8 + i][r] = e1[i];
    }
    __syncthreads();

    BF16x16 afrag[2], bfrag[2];
    #pragma unroll
    for (int tm = 0; tm < 2; tm++) {
      int mb = wm * 32 + tm * 16 + lm;
      afrag[tm].q[0] = *reinterpret_cast<const u32x4*>(&As[mb][lh * 8]);
      afrag[tm].q[1] = *reinterpret_cast<const u32x4*>(&As[mb][16 + lh * 8]);
    }
    #pragma unroll
    for (int tn = 0; tn < 2; tn++) {
      int nb = wn * 32 + tn * 16 + lm;
      bfrag[tn].q[0] = *reinterpret_cast<const u32x4*>(&Bst[nb][lh * 8]);
      bfrag[tn].q[1] = *reinterpret_cast<const u32x4*>(&Bst[nb][16 + lh * 8]);
    }
    #pragma unroll
    for (int tm = 0; tm < 2; tm++)
      #pragma unroll
      for (int tn = 0; tn < 2; tn++)
        acc[tm][tn] = wmma_bf16(afrag[tm], bfrag[tn], acc[tm][tn]);
  }

  // Epilogue: C lane layout n=lane%16, m=(lane/16)*8 + r
  #pragma unroll
  for (int tm = 0; tm < 2; tm++) {
    #pragma unroll
    for (int tn = 0; tn < 2; tn++) {
      int gn = n0 + wn * 32 + tn * 16 + lm;
      #pragma unroll
      for (int r = 0; r < 8; r++) {
        int gm = m0 + wm * 32 + tm * 16 + lh * 8 + r;
        if (gm < Mdim) {
          float val = acc[tm][tn][r];
          if (bias)     val += bias[gn];
          if (relu)     val = fmaxf(val, 0.0f);
          if (residual) val += residual[(size_t)gm * Ndim + gn];
          C[(size_t)gm * Ndim + gn] = val;
        }
      }
    }
  }
}

// =====================================================================
// Flash attention: per block = 128 queries of one (b,head); wave owns 16 q.
// Q/K/V bf16 [M, HID] with head slice at col head*32. Output fp32 [M, HID].
// =====================================================================
__global__ __launch_bounds__(256)
void k_attn(const u16* __restrict__ Q, const u16* __restrict__ K,
            const u16* __restrict__ Vm, float* __restrict__ O)
{
  __shared__ u16 Ks [32][40];      // [key][d]
  __shared__ u16 Vst[32][40];      // [d][key] (transposed)
  __shared__ u16 Ps [8][16][40];   // per-wave P tile [m][k]

  const int tid = threadIdx.x, wave = tid >> 5, lane = tid & 31;
  const int lm = lane & 15, lh = lane >> 4;
  const int b  = blockIdx.y / HEADS_, hd = blockIdx.y % HEADS_;
  const int coff = hd * DH_;
  const int q0 = blockIdx.x * 128 + wave * 16;
  const size_t rowbase = (size_t)b * V_;

  BF16x16 qa;  // Q fragment (A layout), loaded once
  {
    size_t qr = (rowbase + q0 + lm) * HID_ + coff;
    qa.q[0] = *reinterpret_cast<const u32x4*>(Q + qr + lh * 8);
    qa.q[1] = *reinterpret_cast<const u32x4*>(Q + qr + 16 + lh * 8);
  }

  float mrow[8], lsum[8];
  v8f o0, o1;
  #pragma unroll
  for (int r = 0; r < 8; r++) { mrow[r] = -1e30f; lsum[r] = 0.f; o0[r] = 0.f; o1[r] = 0.f; }
  const float scale = 0.17677669529663687f;  // 1/sqrt(32)

  for (int kt = 0; kt < V_; kt += 32) {
    __syncthreads();
    { // stage 32 keys x 32 dims of K and V
      int idx = tid * 4;
      int r = idx >> 5, c = idx & 31;
      size_t gr = (rowbase + kt + r) * HID_ + coff + c;
      u32x2 kv = *reinterpret_cast<const u32x2*>(K + gr);
      *reinterpret_cast<u32x2*>(&Ks[r][c]) = kv;
      u32x2 vv = *reinterpret_cast<const u32x2*>(Vm + gr);
      const u16* vp = reinterpret_cast<const u16*>(&vv);
      #pragma unroll
      for (int i = 0; i < 4; i++) Vst[c + i][r] = vp[i];
    }
    __syncthreads();

    // S = Q @ K^T : Bmat[d][key] = Ks[key][d]
    BF16x16 kb0, kb1;
    kb0.q[0] = *reinterpret_cast<const u32x4*>(&Ks[lm][lh * 8]);
    kb0.q[1] = *reinterpret_cast<const u32x4*>(&Ks[lm][16 + lh * 8]);
    kb1.q[0] = *reinterpret_cast<const u32x4*>(&Ks[16 + lm][lh * 8]);
    kb1.q[1] = *reinterpret_cast<const u32x4*>(&Ks[16 + lm][16 + lh * 8]);
    v8f s0, s1;
    #pragma unroll
    for (int r = 0; r < 8; r++) { s0[r] = 0.f; s1[r] = 0.f; }
    s0 = wmma_bf16(qa, kb0, s0);
    s1 = wmma_bf16(qa, kb1, s1);

    // online softmax (row m = lh*8+r lives in the lane's 16-lane half)
    #pragma unroll
    for (int r = 0; r < 8; r++) {
      float x0 = s0[r] * scale, x1 = s1[r] * scale;
      float mx = fmaxf(x0, x1);
      #pragma unroll
      for (int d = 1; d < 16; d <<= 1) mx = fmaxf(mx, __shfl_xor(mx, d, 32));
      float newm  = fmaxf(mrow[r], mx);
      float alpha = __expf(mrow[r] - newm);
      float p0 = __expf(x0 - newm), p1 = __expf(x1 - newm);
      float ts = p0 + p1;
      #pragma unroll
      for (int d = 1; d < 16; d <<= 1) ts += __shfl_xor(ts, d, 32);
      lsum[r] = lsum[r] * alpha + ts;
      mrow[r] = newm;
      o0[r] *= alpha; o1[r] *= alpha;
      Ps[wave][lh * 8 + r][lm]      = f2bf(p0);
      Ps[wave][lh * 8 + r][16 + lm] = f2bf(p1);
    }

    // O += P @ V : P re-read in A layout (wave-local LDS, in-order per wave)
    BF16x16 pa, vb0, vb1;
    pa.q[0] = *reinterpret_cast<const u32x4*>(&Ps[wave][lm][lh * 8]);
    pa.q[1] = *reinterpret_cast<const u32x4*>(&Ps[wave][lm][16 + lh * 8]);
    vb0.q[0] = *reinterpret_cast<const u32x4*>(&Vst[lm][lh * 8]);
    vb0.q[1] = *reinterpret_cast<const u32x4*>(&Vst[lm][16 + lh * 8]);
    vb1.q[0] = *reinterpret_cast<const u32x4*>(&Vst[16 + lm][lh * 8]);
    vb1.q[1] = *reinterpret_cast<const u32x4*>(&Vst[16 + lm][16 + lh * 8]);
    o0 = wmma_bf16(pa, vb0, o0);
    o1 = wmma_bf16(pa, vb1, o1);
  }

  #pragma unroll
  for (int r = 0; r < 8; r++) {
    float inv = 1.0f / lsum[r];
    size_t gr = (rowbase + q0 + lh * 8 + r) * HID_ + coff;
    O[gr + lm]      = o0[r] * inv;
    O[gr + 16 + lm] = o1[r] * inv;
  }
}

// ---------------- norms ----------------
__global__ __launch_bounds__(256)
void k_lnorm(const float* __restrict__ X, const float* __restrict__ gw,
             const float* __restrict__ bw, float* __restrict__ Y)
{
  int wave = threadIdx.x >> 5, lane = threadIdx.x & 31;
  size_t row = (size_t)blockIdx.x * 8 + wave;
  const float* x = X + row * HID_;
  float s = 0.f;
  for (int c = lane; c < HID_; c += 32) s += x[c];
  #pragma unroll
  for (int d = 1; d < 32; d <<= 1) s += __shfl_xor(s, d, 32);
  float mu = s * (1.0f / HID_);
  float v = 0.f;
  for (int c = lane; c < HID_; c += 32) { float t = x[c] - mu; v += t * t; }
  #pragma unroll
  for (int d = 1; d < 32; d <<= 1) v += __shfl_xor(v, d, 32);
  float inv = rsqrtf(v * (1.0f / HID_) + 1e-5f);
  float* y = Y + row * HID_;
  for (int c = lane; c < HID_; c += 32) y[c] = (x[c] - mu) * inv * gw[c] + bw[c];
}

__global__ __launch_bounds__(256)
void k_gnorm(const float* __restrict__ X, const float* __restrict__ gg,
             const float* __restrict__ gb, float* __restrict__ Y)
{
  __shared__ float r1[256], r2[256];
  __shared__ float mu_s, inv_s;
  const int n = V_ * HID_;
  int b = blockIdx.x;
  const float* x = X + (size_t)b * n;
  float* y = Y + (size_t)b * n;
  float s = 0.f, ss = 0.f;
  for (int i = threadIdx.x; i < n; i += 256) { float v = x[i]; s += v; ss += v * v; }
  r1[threadIdx.x] = s; r2[threadIdx.x] = ss;
  __syncthreads();
  for (int st = 128; st > 0; st >>= 1) {
    if (threadIdx.x < st) { r1[threadIdx.x] += r1[threadIdx.x + st]; r2[threadIdx.x] += r2[threadIdx.x + st]; }
    __syncthreads();
  }
  if (threadIdx.x == 0) {
    float mu = r1[0] / n;
    float var = r2[0] / n - mu * mu;
    mu_s = mu; inv_s = rsqrtf(var + 1e-5f);
  }
  __syncthreads();
  float mu = mu_s, inv = inv_s;
  for (int i = threadIdx.x; i < n; i += 256) {
    int c = i & (HID_ - 1);
    y[i] = (x[i] - mu) * inv * gg[c] + gb[c];
  }
}

// ---------------- graph / elementwise ----------------
__global__ void k_fill(float* p, int n, float v) {
  int i = blockIdx.x * 256 + threadIdx.x; if (i < n) p[i] = v;
}
__global__ void k_degacc(const int* __restrict__ dst, float* deg, int E) {
  int i = blockIdx.x * 256 + threadIdx.x; if (i < E) atomicAdd(&deg[dst[i]], 1.0f);
}
__global__ void k_edgenorm(const int* __restrict__ src, const int* __restrict__ dst,
                           const float* __restrict__ deg, float* en, int E) {
  int i = blockIdx.x * 256 + threadIdx.x;
  if (i < E) en[i] = rsqrtf(deg[src[i]]) * rsqrtf(deg[dst[i]]);
}
__global__ void k_snorm(const float* deg, float* sn, int n) {
  int i = blockIdx.x * 256 + threadIdx.x; if (i < n) sn[i] = 1.0f / deg[i];
}
__global__ void k_cast(const float* __restrict__ s, u16* __restrict__ d, size_t n) {
  size_t i = (size_t)blockIdx.x * 256 + threadIdx.x; if (i < n) d[i] = f2bf(s[i]);
}
__global__ void k_concat(const float* __restrict__ x, const float* __restrict__ cond,
                         u16* __restrict__ A) {
  const int W = FIN_ + COND_;
  size_t i = (size_t)blockIdx.x * 256 + threadIdx.x;
  if (i >= (size_t)M_ * W) return;
  size_t row = i / W; int c = (int)(i % W);
  int v = (int)(row % V_);
  float val = (c < FIN_) ? x[row * FIN_ + c] : cond[(size_t)v * COND_ + (c - FIN_)];
  A[i] = f2bf(val);
}
__global__ void k_gcn_combine(const float* __restrict__ hw, const float* __restrict__ sn,
                              const float* __restrict__ bg, float* __restrict__ out) {
  size_t i = (size_t)blockIdx.x * 256 + threadIdx.x;
  if (i >= (size_t)M_ * HID_) return;
  int v = (int)((i >> 8) % V_); int c = (int)(i & 255);
  out[i] = hw[i] * sn[v] + bg[c];
}
__global__ __launch_bounds__(256)
void k_scatter(const float* __restrict__ hw, const int* __restrict__ src,
               const int* __restrict__ dst, const float* __restrict__ en,
               float* __restrict__ out) {
  int e = blockIdx.x, c = threadIdx.x;
  int s = src[e], d = dst[e]; float w = en[e];
  #pragma unroll
  for (int b = 0; b < B_; b++)
    atomicAdd(&out[((size_t)b * V_ + d) * HID_ + c],
              hw[((size_t)b * V_ + s) * HID_ + c] * w);
}
__global__ void k_relu_ce(const float* __restrict__ g, const float* __restrict__ ce,
                          float* __restrict__ h) {
  size_t i = (size_t)blockIdx.x * 256 + threadIdx.x;
  if (i >= (size_t)M_ * HID_) return;
  int v = (int)((i >> 8) % V_); int c = (int)(i & 255);
  h[i] = fmaxf(g[i], 0.0f) + ce[(size_t)v * HID_ + c];
}
__global__ void k_film(const float* __restrict__ g, const float* __restrict__ tf,
                       const float* __restrict__ ce, float* __restrict__ h) {
  size_t i = (size_t)blockIdx.x * 256 + threadIdx.x;
  if (i >= (size_t)M_ * HID_) return;
  int c = (int)(i & 255);
  size_t row = i >> 8;
  int v = (int)(row % V_); int b = (int)(row / V_);
  float val = g[i] * tf[b * 2 * HID_ + c] + tf[b * 2 * HID_ + HID_ + c];
  h[i] = fmaxf(val, 0.0f) + ce[(size_t)v * HID_ + c];
}

// =====================================================================
extern "C" void kernel_launch(void* const* d_in, const int* in_sizes, int n_in,
                              void* d_out, int out_size, void* d_ws, size_t ws_size,
                              hipStream_t stream) {
  (void)n_in; (void)out_size; (void)ws_size;
  const float* x      = (const float*)d_in[0];
  const float* cond_x = (const float*)d_in[1];
  const int*   edge   = (const int*)d_in[2];
  const float* tvec   = (const float*)d_in[3];
  const int E = in_sizes[2] / 2;
  const int* src = edge;
  const int* dst = edge + E;
  // params pytree (sorted dict keys): Wfilm, bfilm, gn_b, gn_g, layers[...]
  const float* Wfilm = (const float*)d_in[4];
  const float* bfilm = (const float*)d_in[5];
  const float* gn_b  = (const float*)d_in[6];
  const float* gn_g  = (const float*)d_in[7];
  // per-layer sorted leaves: We Wf1 Wf2 Wg Wk Wl Wo Wq Wv be bf1 bf2 bg bk bl bo bq bv ln_b ln_g
  auto L = [&](int i, int j) -> const float* { return (const float*)d_in[8 + 20 * i + j]; };

  // -------- workspace bump allocator --------
  char* base = (char*)d_ws;
  size_t off = 0;
  auto alloc = [&](size_t bytes) -> void* {
    void* p = base + off;
    off += (bytes + 255) & ~(size_t)255;
    return p;
  };
  float* deg   = (float*)alloc((size_t)V_ * 4);
  float* enorm = (float*)alloc((size_t)E * 4);
  float* snorm = (float*)alloc((size_t)V_ * 4);
  float* ce    = (float*)alloc((size_t)V_ * HID_ * 4);
  float* h     = (float*)alloc((size_t)M_ * HID_ * 4);
  float* g     = (float*)alloc((size_t)M_ * HID_ * 4);
  float* t32   = (float*)alloc((size_t)M_ * FF_ * 4);
  float* tf    = (float*)alloc((size_t)B_ * 2 * HID_ * 4);
  u16* Abf   = (u16*)alloc((size_t)M_ * FF_ * 2);
  u16* qb    = (u16*)alloc((size_t)M_ * HID_ * 2);
  u16* kb    = (u16*)alloc((size_t)M_ * HID_ * 2);
  u16* vb    = (u16*)alloc((size_t)M_ * HID_ * 2);
  u16* condb = (u16*)alloc((size_t)V_ * COND_ * 2);
  u16* tb    = (u16*)alloc((size_t)B_ * ENC_ * 2);
  u16 *WgB[NL_], *WeB[NL_], *WqB[NL_], *WkB[NL_], *WvB[NL_], *WoB[NL_],
      *Wf1B[NL_], *Wf2B[NL_], *WlB[NL_], *WfilmB;
  for (int i = 0; i < NL_; i++) {
    int fi = (i == 0) ? (FIN_ + COND_) : HID_;
    WgB[i]  = (u16*)alloc((size_t)fi * HID_ * 2);
    WeB[i]  = (u16*)alloc((size_t)COND_ * HID_ * 2);
    WqB[i]  = (u16*)alloc((size_t)HID_ * HID_ * 2);
    WkB[i]  = (u16*)alloc((size_t)HID_ * HID_ * 2);
    WvB[i]  = (u16*)alloc((size_t)HID_ * HID_ * 2);
    WoB[i]  = (u16*)alloc((size_t)HID_ * HID_ * 2);
    Wf1B[i] = (u16*)alloc((size_t)HID_ * FF_ * 2);
    Wf2B[i] = (u16*)alloc((size_t)FF_ * HID_ * 2);
    WlB[i]  = (u16*)alloc((size_t)HID_ * HID_ * 2);
  }
  WfilmB = (u16*)alloc((size_t)ENC_ * 2 * HID_ * 2);

  auto cast = [&](const float* s, u16* d, size_t n) {
    unsigned blocks = (unsigned)((n + 255) / 256);
    k_cast<<<dim3(blocks), 256, 0, stream>>>(s, d, n);
  };
  auto gemm = [&](const u16* A, const u16* W, const float* bias, const float* resid,
                  float* Cc, int Mm, int Nn, int Kk, int relu) {
    dim3 grid((unsigned)(Nn / 128), (unsigned)((Mm + 63) / 64));
    k_gemm<<<grid, 256, 0, stream>>>(A, W, bias, resid, Cc, Mm, Nn, Kk, relu);
  };
  const size_t MH = (size_t)M_ * HID_;
  unsigned bMH = (unsigned)((MH + 255) / 256);

  // -------- weight / input casts to bf16 --------
  cast(cond_x, condb, (size_t)V_ * COND_);
  cast(tvec, tb, (size_t)B_ * ENC_);
  cast(Wfilm, WfilmB, (size_t)ENC_ * 2 * HID_);
  for (int i = 0; i < NL_; i++) {
    int fi = (i == 0) ? (FIN_ + COND_) : HID_;
    cast(L(i, 3), WgB[i],  (size_t)fi * HID_);
    cast(L(i, 0), WeB[i],  (size_t)COND_ * HID_);
    cast(L(i, 7), WqB[i],  (size_t)HID_ * HID_);
    cast(L(i, 4), WkB[i],  (size_t)HID_ * HID_);
    cast(L(i, 8), WvB[i],  (size_t)HID_ * HID_);
    cast(L(i, 6), WoB[i],  (size_t)HID_ * HID_);
    cast(L(i, 1), Wf1B[i], (size_t)HID_ * FF_);
    cast(L(i, 2), Wf2B[i], (size_t)FF_ * HID_);
    cast(L(i, 5), WlB[i],  (size_t)HID_ * HID_);
  }

  // -------- graph normalization terms --------
  k_fill<<<dim3((V_ + 255) / 256), 256, 0, stream>>>(deg, V_, 1.0f);
  k_degacc<<<dim3((unsigned)((E + 255) / 256)), 256, 0, stream>>>(dst, deg, E);
  k_edgenorm<<<dim3((unsigned)((E + 255) / 256)), 256, 0, stream>>>(src, dst, deg, enorm, E);
  k_snorm<<<dim3((V_ + 255) / 256), 256, 0, stream>>>(deg, snorm, V_);

  // -------- layers --------
  for (int li = 0; li < NL_; li++) {
    int fi = (li == 0) ? (FIN_ + COND_) : HID_;
    // ce = cond @ We + be
    gemm(condb, WeB[li], L(li, 9), nullptr, ce, V_, HID_, COND_, 0);

    // GCN input -> Abf (bf16)
    if (li == 0) {
      size_t n = (size_t)M_ * fi;
      k_concat<<<dim3((unsigned)((n + 255) / 256)), 256, 0, stream>>>(x, cond_x, Abf);
    } else if (li == 1) {
      k_gnorm<<<dim3(B_), 256, 0, stream>>>(h, gn_g, gn_b, g);
      cast(g, Abf, MH);
    } else {
      cast(h, Abf, MH);
    }
    // hW = A @ Wg (no bias); g = hW*snorm + bg; scatter-add edges
    gemm(Abf, WgB[li], nullptr, nullptr, t32, M_, HID_, fi, 0);
    k_gcn_combine<<<dim3(bMH), 256, 0, stream>>>(t32, snorm, L(li, 12), g);
    k_scatter<<<dim3((unsigned)E), 256, 0, stream>>>(t32, src, dst, enorm, g);

    if (li < NL_ - 1) {
      k_relu_ce<<<dim3(bMH), 256, 0, stream>>>(g, ce, h);
    } else {
      gemm(tb, WfilmB, bfilm, nullptr, tf, B_, 2 * HID_, ENC_, 0);
      k_film<<<dim3(bMH), 256, 0, stream>>>(g, tf, ce, h);
    }

    // ---- attention ----
    cast(h, Abf, MH);
    gemm(Abf, WqB[li], L(li, 16), nullptr, t32, M_, HID_, HID_, 0); cast(t32, qb, MH);
    gemm(Abf, WkB[li], L(li, 13), nullptr, t32, M_, HID_, HID_, 0); cast(t32, kb, MH);
    gemm(Abf, WvB[li], L(li, 17), nullptr, t32, M_, HID_, HID_, 0); cast(t32, vb, MH);
    k_attn<<<dim3(V_ / 128, B_ * HEADS_), 256, 0, stream>>>(qb, kb, vb, t32);
    cast(t32, Abf, MH);
    gemm(Abf, WoB[li], L(li, 15), h, g, M_, HID_, HID_, 0);        // g = h + o@Wo + bo
    cast(g, Abf, MH);
    gemm(Abf, Wf1B[li], L(li, 10), nullptr, t32, M_, FF_, HID_, 1); // relu(g@Wf1+bf1)
    cast(t32, Abf, (size_t)M_ * FF_);
    gemm(Abf, Wf2B[li], L(li, 11), g, h, M_, HID_, FF_, 0);        // h = g + ff

    // ---- layer norm + per-layer linear ----
    k_lnorm<<<dim3(M_ / 8), 256, 0, stream>>>(h, L(li, 19), L(li, 18), t32);
    cast(t32, Abf, MH);
    if (li < NL_ - 1)
      gemm(Abf, WlB[li], L(li, 14), nullptr, h, M_, HID_, HID_, 1); // relu
    else
      gemm(Abf, WlB[li], L(li, 14), x, (float*)d_out, M_, HID_, HID_, 0); // + x_skip
  }
}
